// TransferFunctionModel_82042465289177
// MI455X (gfx1250) — matrix operand order repaired and checked by reading, not compiled, not run on hardware
//
#include <hip/hip_runtime.h>
#include <hip/hip_bf16.h>

typedef __attribute__((ext_vector_type(16))) _Float16 v16h;
typedef __attribute__((ext_vector_type(8)))  float    v8f;
typedef __attribute__((ext_vector_type(4)))  unsigned int u32x4;
typedef __attribute__((ext_vector_type(8)))  int      i32x8;
typedef __attribute__((ext_vector_type(4)))  int      i32x4;

#define N_SAMPLES     32768
#define CHANNELS      128
#define N_ATOMS       512
#define ATOM_SIZE     512
#define N_TRANSFERS   64
#define TRANSFER_SIZE 32768
#define N_EVENTS      16
#define BATCH         8
#define BANK_KW       512
#define KW            3
#define NEG_SLOPE     0.01f
#define MAX_DIL       27
#define MAX_W         (32 + 2 * MAX_DIL)   // 86

__device__ __forceinline__ v8f wmma16(v16h a, v16h b, v8f c) {
  // D = A(16x32 f16) * B(32x16 f16) + C(16x16 f32)
  return __builtin_amdgcn_wmma_f32_16x16x32_f16(
      /*neg_a=*/false, a, /*neg_b=*/false, b,
      /*c_mod=*/(short)0, c, /*reuse_a=*/false, /*reuse_b=*/false);
}

// Issue a 2D TDM load: tile [rows x W] (f16), row stride `stride` elements,
// from global `src` into LDS at byte offset `lds_byte`. Caller must be
// wave-uniform; completion via s_wait_tensorcnt.
__device__ __forceinline__ void tdm_load_2d_f16(const _Float16* src,
                                                unsigned lds_byte,
                                                int W, int rows, int stride) {
  unsigned long long ga = (unsigned long long)(uintptr_t)src;
  u32x4 g0 = {};
  g0[0] = 1u;                                   // count = 1 valid descriptor
  g0[1] = lds_byte;                             // lds_addr (bytes)
  g0[2] = (unsigned)(ga & 0xffffffffu);         // global_addr[31:0]
  g0[3] = (unsigned)((ga >> 32) & 0x01ffffffu)  // global_addr[56:32]
          | (2u << 30);                         // type = 2 ("image")
  i32x8 g1 = {};
  g1[0] = (1 << 16);                            // data_size = 1 -> 2 bytes
  // tensor_dim0 (bits 79:48) = large (interior tile: no OOB clipping)
  g1[1] = (int)(0xffffu << 16);                 // dim0[15:0]
  g1[2] = (int)(0x7fffu) | (int)((unsigned)rows << 16);  // dim0[31:16], dim1[15:0]
  g1[3] = (int)((unsigned)(W & 0xffff) << 16);  // dim1[31:16]=0, tile_dim0 = W
  g1[4] = rows;                                 // tile_dim1 = rows, tile_dim2 = 0
  g1[5] = stride;                               // tensor_dim0_stride[31:0]
  g1[6] = 0;                                    // stride[47:32], dim1_stride lo
  g1[7] = 0;
  i32x4 gz = {};
#if defined(__clang_major__) && (__clang_major__ >= 23)
  i32x8 gz8 = {};
  __builtin_amdgcn_tensor_load_to_lds(g0, g1, gz, gz, gz8, 0);
#else
  __builtin_amdgcn_tensor_load_to_lds(g0, g1, gz, gz, 0);
#endif
}

// ---------------------------------------------------------------- utilities
__global__ void k_zero_f32(float* __restrict__ p, int n) {
  int i = blockIdx.x * blockDim.x + threadIdx.x;
  if (i < n) p[i] = 0.f;
}

__global__ void k_f32_to_f16(const float* __restrict__ s, _Float16* __restrict__ d, int n) {
  int i = blockIdx.x * blockDim.x + threadIdx.x;
  if (i < n) d[i] = (_Float16)s[i];
}

// ------------------------------------------------- 1) filterbank conv (WMMA)
// h[b,c,t] = sum_k x[b, t+k-256] * bank[c,k]   (implicit GEMM, K = 512)
__global__ __launch_bounds__(256) void k_bank_conv(
    const float* __restrict__ x, const _Float16* __restrict__ bank16,
    _Float16* __restrict__ h) {
  __shared__ _Float16 xs[BANK_KW + 32];
  const int b   = blockIdx.y;
  const int t0  = blockIdx.x * 32;
  const int tid = threadIdx.x;
  for (int i = tid; i < BANK_KW + 32; i += 256) {
    int g = t0 - BANK_KW / 2 + i;
    xs[i] = (g >= 0 && g < N_SAMPLES) ? (_Float16)x[(size_t)b * N_SAMPLES + g]
                                      : (_Float16)0.f;
  }
  __syncthreads();
  const int lane  = tid & 31;
  const int wave  = tid >> 5;
  const int mtile = wave >> 2;   // 0..1 (time sub-tile)
  const int nw    = wave & 3;    // 0..3 (channel wave)
  const int mrow  = lane & 15;
  const int khalf = lane >> 4;
  const int ncol  = lane & 15;
#pragma unroll
  for (int nt = nw; nt < 8; nt += 4) {
    const int n0 = nt * 16;
    v8f acc = {};
    for (int kk = 0; kk < BANK_KW / 32; ++kk) {
      const int kbase = kk * 32;
      v16h a, bf;
#pragma unroll
      for (int e = 0; e < 16; ++e) {
        int ka = khalf * 8 + (e < 8 ? e : e + 8);              // A 16x32 layout
        a[e] = xs[mtile * 16 + mrow + kbase + ka];
        int kb = khalf * 8 + (e < 8 ? e : e + 8);              // B 32x16 layout
        bf[e] = bank16[(size_t)(n0 + ncol) * BANK_KW + kbase + kb];
      }
      acc = wmma16(a, bf, acc);
    }
#pragma unroll
    for (int v = 0; v < 8; ++v) {
      int t = t0 + mtile * 16 + khalf * 8 + v;
      int c = n0 + ncol;
      h[((size_t)b * CHANNELS + c) * N_SAMPLES + t] = (_Float16)acc[v];
    }
  }
}

// -------------------------------------- 2) dilated residual conv layer (WMMA)
// out = lrelu(conv_d(in, w) + bias) + in ; K = 128*3 = 384
// Input tile [128 ch x (32+2*dil)] staged to LDS by the Tensor Data Mover
// (interior blocks) or a manual zero-padded copy (boundary blocks).
__global__ __launch_bounds__(256) void k_dilated(
    const _Float16* __restrict__ hin, _Float16* __restrict__ hout,
    const _Float16* __restrict__ w16, const float* __restrict__ bias, int dil) {
  __shared__ _Float16 xs[CHANNELS * (MAX_W + 2)];
  const int b   = blockIdx.y;
  const int t0  = blockIdx.x * 32;
  const int tid = threadIdx.x;
  const int W   = 32 + 2 * dil;
  const int tstart = t0 - dil;
  const size_t base = (size_t)b * CHANNELS * N_SAMPLES;
  const bool interior = (tstart >= 0) && (tstart + W <= N_SAMPLES);
  if (interior) {
    if (tid < 32) {
      tdm_load_2d_f16(hin + base + tstart,
                      (unsigned)(uintptr_t)(&xs[0]),
                      W, CHANNELS, N_SAMPLES);
      __builtin_amdgcn_s_wait_tensorcnt(0);
    }
  } else {
    for (int i = tid; i < CHANNELS * W; i += 256) {
      int ci = i / W, tl = i - ci * W;
      int t = tstart + tl;
      xs[i] = (t >= 0 && t < N_SAMPLES)
                  ? hin[base + (size_t)ci * N_SAMPLES + t]
                  : (_Float16)0.f;
    }
  }
  __syncthreads();

  const int lane = tid & 31, wave = tid >> 5;
  const int mtile = wave >> 2, nw = wave & 3;
  const int mrow = lane & 15, khalf = lane >> 4, ncol = lane & 15;
#pragma unroll
  for (int nt = nw; nt < 8; nt += 4) {
    const int n0 = nt * 16;
    v8f acc = {};
    for (int kk = 0; kk < (CHANNELS * KW) / 32; ++kk) {   // 12 K-steps
      const int kbase = kk * 32;
      v16h a, bf;
#pragma unroll
      for (int e = 0; e < 16; ++e) {
        int fa = kbase + khalf * 8 + (e < 8 ? e : e + 8);
        int ci = fa / 3, kt = fa - ci * 3;
        // hin[b, ci, t0 + m + (kt-1)*dil] == xs[ci*W + m + kt*dil]
        a[e] = xs[ci * W + mtile * 16 + mrow + kt * dil];
        int fb = kbase + khalf * 8 + (e < 8 ? e : e + 8);
        bf[e] = w16[(size_t)(n0 + ncol) * (CHANNELS * KW) + fb];
      }
      acc = wmma16(a, bf, acc);
    }
#pragma unroll
    for (int v = 0; v < 8; ++v) {
      int t = t0 + mtile * 16 + khalf * 8 + v;
      int c = n0 + ncol;
      float r = acc[v] + bias[c];
      r = r > 0.f ? r : NEG_SLOPE * r;
      float resid = (float)xs[c * W + dil + mtile * 16 + khalf * 8 + v];
      hout[base + (size_t)c * N_SAMPLES + t] = (_Float16)(r + resid);
    }
  }
}

// --------------------------------------------------- 3) per-example std stats
__global__ void k_std_pass(const _Float16* __restrict__ h, float* __restrict__ sums) {
  __shared__ float ss[256], sq[256];
  const int b = blockIdx.y, tid = threadIdx.x;
  const size_t nelem = (size_t)CHANNELS * N_SAMPLES;
  const _Float16* p = h + (size_t)b * nelem;
  float s = 0.f, q = 0.f;
  for (size_t i = (size_t)blockIdx.x * 256 + tid; i < nelem; i += 256 * 256) {
    float v = (float)p[i]; s += v; q += v * v;
  }
  ss[tid] = s; sq[tid] = q; __syncthreads();
  for (int st = 128; st > 0; st >>= 1) {
    if (tid < st) { ss[tid] += ss[tid + st]; sq[tid] += sq[tid + st]; }
    __syncthreads();
  }
  if (tid == 0) { atomicAdd(&sums[b * 2], ss[0]); atomicAdd(&sums[b * 2 + 1], sq[0]); }
}

__global__ void k_std_final(const float* __restrict__ sums, float* __restrict__ inv_std) {
  int b = threadIdx.x;
  if (b < BATCH) {
    const float n = (float)((size_t)CHANNELS * N_SAMPLES);
    float mean = sums[b * 2] / n;
    float var  = sums[b * 2 + 1] / n - mean * mean;
    inv_std[b] = 1.f / (sqrtf(fmaxf(var, 0.f)) + 1e-8f);
  }
}

// ------------------------------------------------------------- 4) attention
__global__ void k_attn(const _Float16* __restrict__ h, const float* __restrict__ w,
                       const float* __restrict__ bptr, const float* __restrict__ inv_std,
                       float* __restrict__ attn) {
  const int b = blockIdx.y;
  const int t = blockIdx.x * 256 + threadIdx.x;
  const _Float16* p = h + (size_t)b * CHANNELS * N_SAMPLES + t;
  float acc = 0.f;
#pragma unroll 4
  for (int c = 0; c < CHANNELS; ++c) acc += (float)p[(size_t)c * N_SAMPLES] * w[c];
  float z = acc * inv_std[b] + bptr[0];
  attn[(size_t)b * N_SAMPLES + t] = 1.f / (1.f + expf(-z));
}

// ----------------------------------------------- 5) top-16 (iterative argmax)
__global__ void k_topk(float* __restrict__ attn, int* __restrict__ idxs,
                       float* __restrict__ vals) {
  __shared__ float rv[256];
  __shared__ int   ri[256];
  const int b = blockIdx.x, tid = threadIdx.x;
  float* a = attn + (size_t)b * N_SAMPLES;
  for (int e = 0; e < N_EVENTS; ++e) {
    float bv = -1e30f; int bi = 0;
    for (int t = tid; t < N_SAMPLES; t += 256) {
      float v = a[t];
      if (v > bv) { bv = v; bi = t; }
    }
    rv[tid] = bv; ri[tid] = bi; __syncthreads();
    for (int s = 128; s > 0; s >>= 1) {
      if (tid < s && rv[tid + s] > rv[tid]) { rv[tid] = rv[tid + s]; ri[tid] = ri[tid + s]; }
      __syncthreads();
    }
    if (tid == 0) {
      idxs[b * N_EVENTS + e] = ri[0];
      vals[b * N_EVENTS + e] = rv[0];
      a[ri[0]] = -1e30f;
    }
    __syncthreads();
  }
}

// ---------------------------------------------------------- 6) latent gather
__global__ void k_latents(const _Float16* __restrict__ h, const int* __restrict__ idxs,
                          const float* __restrict__ vals, const float* __restrict__ inv_std,
                          _Float16* __restrict__ lat) {
  int ev = blockIdx.x, c = threadIdx.x;  // 128 events x 128 channels
  int b = ev >> 4;
  float sc = vals[ev] * inv_std[b];
  lat[ev * CHANNELS + c] =
      (_Float16)((float)h[((size_t)b * CHANNELS + c) * N_SAMPLES + idxs[ev]] * sc);
}

// ------------------------------------- 7) MLP GEMM: [128 x 128] x [128 x N]
__global__ void k_gemm128(const _Float16* __restrict__ A, const _Float16* __restrict__ W,
                          const float* __restrict__ bias, float* __restrict__ outF,
                          _Float16* __restrict__ outH, int N, int relu) {
  const int mtile = blockIdx.x;          // 0..7
  const int n0    = blockIdx.y * 16;
  const int lane  = threadIdx.x & 31;
  const int mrow = lane & 15, khalf = lane >> 4, ncol = lane & 15;
  const int n = n0 + ncol;
  v8f acc = {};
  for (int kk = 0; kk < 4; ++kk) {
    int kbase = kk * 32;
    v16h a, bf;
#pragma unroll
    for (int e = 0; e < 16; ++e) {
      int ka = kbase + khalf * 8 + (e < 8 ? e : e + 8);
      a[e] = A[(mtile * 16 + mrow) * 128 + ka];
      int kb = kbase + khalf * 8 + (e < 8 ? e : e + 8);
      bf[e] = (n < N) ? W[kb * N + n] : (_Float16)0.f;
    }
    acc = wmma16(a, bf, acc);
  }
  if (n < N) {
#pragma unroll
    for (int v = 0; v < 8; ++v) {
      int m = mtile * 16 + khalf * 8 + v;
      float r = acc[v] + bias[n];
      if (relu) r = r > 0.f ? r : NEG_SLOPE * r;
      if (outF) outF[m * N + n] = r;
      else      outH[m * N + n] = (_Float16)r;
    }
  }
}

// ------------------------------------------------- 8) max-norm denominators
__global__ void k_rowmax(const float* __restrict__ src, float* __restrict__ norms,
                         int rowlen) {
  __shared__ float rv[256];
  int row = blockIdx.x, tid = threadIdx.x;
  const float* p = src + (size_t)row * rowlen;
  float m = 0.f;
  for (int i = tid; i < rowlen; i += 256) m = fmaxf(m, fabsf(p[i]));
  rv[tid] = m; __syncthreads();
  for (int s = 128; s > 0; s >>= 1) {
    if (tid < s) rv[tid] = fmaxf(rv[tid], rv[tid + s]);
    __syncthreads();
  }
  if (tid == 0) norms[row] = rv[0] + 1e-8f;
}

// --------------------------------- 9) per-event argmax heads + amp sigmoid
__global__ void k_heads(const float* __restrict__ lA, const float* __restrict__ lT,
                        const float* __restrict__ lP, int* __restrict__ selA,
                        int* __restrict__ selT, float* __restrict__ amp) {
  __shared__ float rv[256];
  __shared__ int   ri[256];
  const int ev = blockIdx.x, tid = threadIdx.x;
  // atoms argmax over 512
  float bv = -1e30f; int bi = 0;
  for (int i = tid; i < N_ATOMS; i += 256) {
    float v = lA[(size_t)ev * N_ATOMS + i];
    if (v > bv) { bv = v; bi = i; }
  }
  rv[tid] = bv; ri[tid] = bi; __syncthreads();
  for (int s = 128; s > 0; s >>= 1) {
    if (tid < s && rv[tid + s] > rv[tid]) { rv[tid] = rv[tid + s]; ri[tid] = ri[tid + s]; }
    __syncthreads();
  }
  if (tid == 0) selA[ev] = ri[0];
  __syncthreads();
  // transfer argmax over 64
  bv = -1e30f; bi = 0;
  if (tid < N_TRANSFERS) { bv = lT[ev * N_TRANSFERS + tid]; bi = tid; }
  rv[tid] = bv; ri[tid] = bi; __syncthreads();
  for (int s = 128; s > 0; s >>= 1) {
    if (tid < s && rv[tid + s] > rv[tid]) { rv[tid] = rv[tid + s]; ri[tid] = ri[tid + s]; }
    __syncthreads();
  }
  if (tid == 0) {
    selT[ev] = ri[0];
    amp[ev]  = 1.f / (1.f + expf(-lP[ev]));
  }
}

// ------------------- 10) event synth: 512-tap FIR + atom add + atomic scatter
__global__ __launch_bounds__(256) void k_synth(
    const float* __restrict__ impulse, const float* __restrict__ transfer,
    const float* __restrict__ imp_norm, const float* __restrict__ tr_norm,
    const int* __restrict__ selA, const int* __restrict__ selT,
    const float* __restrict__ amp, const int* __restrict__ idxs,
    float* __restrict__ buf) {
  __shared__ float atom_s[ATOM_SIZE];
  __shared__ float tr_s[ATOM_SIZE + 256];
  const int ev = blockIdx.y;
  const int t0 = blockIdx.x * 256;
  const int tid = threadIdx.x;
  const int sa = selA[ev], st = selT[ev];
  const float asc = amp[ev] / imp_norm[sa];
  const float tsc = 1.f / tr_norm[st];
  for (int i = tid; i < ATOM_SIZE; i += 256)
    atom_s[i] = impulse[(size_t)sa * ATOM_SIZE + i] * asc;
  for (int i = tid; i < ATOM_SIZE + 256; i += 256) {
    int j = t0 - (ATOM_SIZE - 1) + i;
    tr_s[i] = (j >= 0 && j < TRANSFER_SIZE)
                  ? transfer[(size_t)st * TRANSFER_SIZE + j] * tsc : 0.f;
  }
  __syncthreads();
  const int t = t0 + tid;
  float acc = (t < ATOM_SIZE) ? atom_s[t] : 0.f;  // "+ atoms" term
#pragma unroll 8
  for (int k = 0; k < ATOM_SIZE; ++k)
    acc += atom_s[k] * tr_s[tid + (ATOM_SIZE - 1) - k];
  atomicAdd(&buf[idxs[ev] + t], acc);
}

// ----------------------------------------------------- 11) broadcast output
__global__ void k_out(const float* __restrict__ buf, float* __restrict__ out) {
  int i = blockIdx.x * 256 + threadIdx.x;
  if (i < BATCH * N_SAMPLES) out[i] = buf[i & (N_SAMPLES - 1)];
}

// ===========================================================================
extern "C" void kernel_launch(void* const* d_in, const int* in_sizes, int n_in,
                              void* d_out, int out_size, void* d_ws, size_t ws_size,
                              hipStream_t stream) {
  (void)in_sizes; (void)n_in; (void)out_size; (void)ws_size;
  const float* x      = (const float*)d_in[0];
  const float* bank   = (const float*)d_in[1];
  const float* netW[5]; const float* netB[5];
  for (int i = 0; i < 5; ++i) {
    netW[i] = (const float*)d_in[2 + 2 * i];
    netB[i] = (const float*)d_in[3 + 2 * i];
  }
  const float* attn_w = (const float*)d_in[12];
  const float* attn_b = (const float*)d_in[13];
  const float *aW[3], *aB[3], *tW[3], *tB[3], *pW[3], *pB[3];
  for (int i = 0; i < 3; ++i) {
    aW[i] = (const float*)d_in[14 + 2 * i]; aB[i] = (const float*)d_in[15 + 2 * i];
    tW[i] = (const float*)d_in[20 + 2 * i]; tB[i] = (const float*)d_in[21 + 2 * i];
    pW[i] = (const float*)d_in[26 + 2 * i]; pB[i] = (const float*)d_in[27 + 2 * i];
  }
  const float* impulse  = (const float*)d_in[32];
  const float* transfer = (const float*)d_in[33];
  float* out = (float*)d_out;

  // ---- workspace bump allocator (256B aligned)
  char* p = (char*)d_ws;
  auto bump = [&](size_t bytes) -> void* {
    void* r = (void*)p;
    p += (bytes + 255) & ~(size_t)255;
    return r;
  };
  const size_t HN = (size_t)BATCH * CHANNELS * N_SAMPLES;
  _Float16* hA     = (_Float16*)bump(HN * 2);
  _Float16* hB     = (_Float16*)bump(HN * 2);
  _Float16* bank16 = (_Float16*)bump((size_t)CHANNELS * BANK_KW * 2);
  _Float16* w16[5];
  for (int i = 0; i < 5; ++i) w16[i] = (_Float16*)bump((size_t)CHANNELS * CHANNELS * KW * 2);
  _Float16* aWh[3]; _Float16* tWh[3]; _Float16* pWh[3];
  int aN[3] = {128, 128, N_ATOMS}, tN[3] = {128, 128, N_TRANSFERS}, pN[3] = {128, 128, 1};
  for (int i = 0; i < 3; ++i) aWh[i] = (_Float16*)bump((size_t)128 * aN[i] * 2);
  for (int i = 0; i < 3; ++i) tWh[i] = (_Float16*)bump((size_t)128 * tN[i] * 2);
  for (int i = 0; i < 3; ++i) pWh[i] = (_Float16*)bump((size_t)128 * pN[i] * 2);
  float* sums     = (float*)bump(16 * 4);
  float* inv_std  = (float*)bump(BATCH * 4);
  float* attn     = (float*)bump((size_t)BATCH * N_SAMPLES * 4);
  int*   idxs     = (int*)bump(128 * 4);
  float* vals     = (float*)bump(128 * 4);
  _Float16* lat16 = (_Float16*)bump(128 * 128 * 2);
  _Float16* tmp0  = (_Float16*)bump(128 * 128 * 2);
  _Float16* tmp1  = (_Float16*)bump(128 * 128 * 2);
  float* logitsA  = (float*)bump((size_t)128 * N_ATOMS * 4);
  float* logitsT  = (float*)bump((size_t)128 * N_TRANSFERS * 4);
  float* logitsP  = (float*)bump(128 * 4);
  float* imp_norm = (float*)bump(N_ATOMS * 4);
  float* tr_norm  = (float*)bump(N_TRANSFERS * 4);
  float* buf      = (float*)bump((size_t)2 * N_SAMPLES * 4);
  int*   selA     = (int*)bump(128 * 4);
  int*   selT     = (int*)bump(128 * 4);
  float* ampv     = (float*)bump(128 * 4);

  // ---- zero accumulators (required every call: buf/sums are atomically added)
  k_zero_f32<<<dim3((2 * N_SAMPLES + 255) / 256), 256, 0, stream>>>(buf, 2 * N_SAMPLES);
  k_zero_f32<<<1, 32, 0, stream>>>(sums, 16);

  // ---- weight conversion f32 -> f16
  auto cvt = [&](const float* s, _Float16* d, int n) {
    k_f32_to_f16<<<dim3((n + 255) / 256), 256, 0, stream>>>(s, d, n);
  };
  cvt(bank, bank16, CHANNELS * BANK_KW);
  for (int i = 0; i < 5; ++i) cvt(netW[i], w16[i], CHANNELS * CHANNELS * KW);
  for (int i = 0; i < 3; ++i) cvt(aW[i], aWh[i], 128 * aN[i]);
  for (int i = 0; i < 3; ++i) cvt(tW[i], tWh[i], 128 * tN[i]);
  for (int i = 0; i < 3; ++i) cvt(pW[i], pWh[i], 128 * pN[i]);

  // ---- analysis filterbank (WMMA implicit GEMM)
  k_bank_conv<<<dim3(N_SAMPLES / 32, BATCH), 256, 0, stream>>>(x, bank16, hA);

  // ---- dilated residual stack (WMMA + TDM tile staging), ping-pong hA<->hB
  const int dils[5] = {1, 3, 9, 27, 1};
  _Float16* cur = hA; _Float16* nxt = hB;
  for (int i = 0; i < 5; ++i) {
    k_dilated<<<dim3(N_SAMPLES / 32, BATCH), 256, 0, stream>>>(cur, nxt, w16[i], netB[i], dils[i]);
    _Float16* t = cur; cur = nxt; nxt = t;
  }

  // ---- per-example std, attention, top-k, latents
  k_std_pass<<<dim3(256, BATCH), 256, 0, stream>>>(cur, sums);
  k_std_final<<<1, 32, 0, stream>>>(sums, inv_std);
  k_attn<<<dim3(N_SAMPLES / 256, BATCH), 256, 0, stream>>>(cur, attn_w, attn_b, inv_std, attn);
  k_topk<<<BATCH, 256, 0, stream>>>(attn, idxs, vals);
  k_latents<<<128, 128, 0, stream>>>(cur, idxs, vals, inv_std, lat16);

  // ---- event MLP heads (WMMA GEMMs, M=128 events, K=128)
  auto gemm = [&](const _Float16* A, const _Float16* W, const float* b,
                  float* oF, _Float16* oH, int N, int relu) {
    k_gemm128<<<dim3(8, (N + 15) / 16), 32, 0, stream>>>(A, W, b, oF, oH, N, relu);
  };
  gemm(lat16, aWh[0], aB[0], nullptr, tmp0, 128, 1);
  gemm(tmp0,  aWh[1], aB[1], nullptr, tmp1, 128, 1);
  gemm(tmp1,  aWh[2], aB[2], logitsA, nullptr, N_ATOMS, 0);
  gemm(lat16, tWh[0], tB[0], nullptr, tmp0, 128, 1);
  gemm(tmp0,  tWh[1], tB[1], nullptr, tmp1, 128, 1);
  gemm(tmp1,  tWh[2], tB[2], logitsT, nullptr, N_TRANSFERS, 0);
  gemm(lat16, pWh[0], pB[0], nullptr, tmp0, 128, 1);
  gemm(tmp0,  pWh[1], pB[1], nullptr, tmp1, 128, 1);
  gemm(tmp1,  pWh[2], pB[2], logitsP, nullptr, 1, 0);

  // ---- dictionary max-norms, argmax selection, amp
  k_rowmax<<<N_ATOMS, 256, 0, stream>>>(impulse, imp_norm, ATOM_SIZE);
  k_rowmax<<<N_TRANSFERS, 256, 0, stream>>>(transfer, tr_norm, TRANSFER_SIZE);
  k_heads<<<128, 256, 0, stream>>>(logitsA, logitsT, logitsP, selA, selT, ampv);

  // ---- event synthesis (512-tap FIR) + atomic scatter-add into shared buffer
  k_synth<<<dim3(TRANSFER_SIZE / 256, 128), 256, 0, stream>>>(
      impulse, transfer, imp_norm, tr_norm, selA, selT, ampv, idxs, buf);

  // ---- broadcast to all batch rows
  k_out<<<dim3((BATCH * N_SAMPLES + 255) / 256), 256, 0, stream>>>(buf, out);
}
